// Model_63556926046489
// MI455X (gfx1250) — matrix operand order repaired and checked by reading, not compiled, not run on hardware
//
#include <hip/hip_runtime.h>

// ---------------- problem constants (from reference) ----------------
#define Bc   2
#define Lc   24
#define Kc   16
#define Nc   64
#define Dc   1024
#define Hc   16
#define HDc  64
#define Fc   2816
#define NLc  2
#define Ec   (Kc + Nc)        // 80
#define Sc   (Lc * Ec)        // 1920
#define Tc   (Bc * Sc)        // 3840
#define ThreeD (3 * Dc)       // 3072
#define TwoF   (2 * Fc)       // 5632
#define EPSc 1e-5f

// ---------------- CDNA5 WMMA vector types ----------------
typedef __attribute__((ext_vector_type(16))) __bf16 v16bf;
typedef __attribute__((ext_vector_type(8)))  float  v8f;
typedef __attribute__((ext_vector_type(2)))  int    v2i;

union V16BF { unsigned int u[8]; v16bf v; };

__device__ __forceinline__ unsigned short f2bf(float f) {
    unsigned int u = __float_as_uint(f);
    u += 0x7fffu + ((u >> 16) & 1u);     // round-to-nearest-even
    return (unsigned short)(u >> 16);
}

// ---------------- gfx1250 async global->LDS copy helpers ----------------
__device__ __forceinline__ void async_cp_b64(const unsigned short* g, unsigned short* l) {
#if __has_builtin(__builtin_amdgcn_global_load_async_to_lds_b64)
    __builtin_amdgcn_global_load_async_to_lds_b64(
        (__attribute__((address_space(1))) v2i*)g,
        (__attribute__((address_space(3))) v2i*)l, 0, 0);
#else
    *(unsigned long long*)l = *(const unsigned long long*)g;
#endif
}

__device__ __forceinline__ void wait_async0() {
#if __has_builtin(__builtin_amdgcn_s_wait_asynccnt)
    __builtin_amdgcn_s_wait_asynccnt(0);
#else
    asm volatile("s_wait_asynccnt 0x0" ::: "memory");
#endif
}

// ---------------- generic fp32 -> bf16 convert ----------------
__global__ void conv_bf16_k(const float* __restrict__ X,
                            unsigned short* __restrict__ Y, long n) {
    long i = (long)blockIdx.x * blockDim.x + threadIdx.x;
    if (i < n) Y[i] = f2bf(X[i]);
}

// ---------------- fp32 copy ----------------
__global__ void copy_f32_k(const float* __restrict__ X, float* __restrict__ Y, long n) {
    long i = (long)blockIdx.x * blockDim.x + threadIdx.x;
    if (i < n) Y[i] = X[i];
}

// ---------------- RoPE tables: cos/sin (S x 32) ----------------
__global__ void rope_tables_k(float* __restrict__ cosT, float* __restrict__ sinT) {
    int i = blockIdx.x * blockDim.x + threadIdx.x;
    if (i >= Sc * 32) return;
    int t = i >> 5, j = i & 31;
    float inv = powf(10000.0f, -(2.0f * (float)j) / 64.0f);
    float fr = (float)t * inv;
    cosT[i] = cosf(fr);
    sinT[i] = sinf(fr);
}

// ---------------- RMSNorm -> bf16 (row stride Dc) ----------------
__global__ void rmsnorm_bf16_k(const float* __restrict__ X, const float* __restrict__ g,
                               unsigned short* __restrict__ Obf) {
    __shared__ float red[256];
    const int row = blockIdx.x;
    const float* xr = X + (size_t)row * Dc;
    float s = 0.f;
    for (int i = threadIdx.x; i < Dc; i += 256) { float v = xr[i]; s += v * v; }
    red[threadIdx.x] = s;
    __syncthreads();
    for (int off = 128; off > 0; off >>= 1) {
        if ((int)threadIdx.x < off) red[threadIdx.x] += red[threadIdx.x + off];
        __syncthreads();
    }
    const float scale = rsqrtf(red[0] * (1.0f / Dc) + EPSc);
    for (int i = threadIdx.x; i < Dc; i += 256)
        Obf[(size_t)row * Dc + i] = f2bf(xr[i] * scale * g[i]);
}

// ---------------- WMMA bf16 GEMM: Out[T,N] = A[T,K] * W[K,N] (+Res) ----------------
// Block = 256 threads (8 waves). Block tile: 32 rows x 512 cols.
// Each wave: 32x64 macro-tile = 2 M-subtiles x 4 N-subtiles = 8 WMMAs / K-step.
// Shared A tile (32 rows x 32 K, bf16 = 2KB) staged via async global->LDS copy.
__global__ __launch_bounds__(256)
void gemm_wmma_bf16_k(const unsigned short* __restrict__ A,
                      const unsigned short* __restrict__ Bw,
                      const float* __restrict__ Res,
                      float* __restrict__ Out,
                      int Kdim, int Ndim) {
    __shared__ unsigned short shA[32 * 32];   // [row][k], row stride 32 elems = 64B

    const int tid  = threadIdx.x;             // 0..255
    const int lane = tid & 31;
    const int wave = tid >> 5;                // 0..7
    const int m0   = blockIdx.y * 32;
    const int n0   = blockIdx.x * 512 + wave * 64;
    const int half = lane >> 4;               // 0 or 1
    const int l16  = lane & 15;

    v8f c[2][4];
    #pragma unroll
    for (int mt = 0; mt < 2; ++mt)
        #pragma unroll
        for (int nt = 0; nt < 4; ++nt) { v8f z = {}; c[mt][nt] = z; }

    // A staging: each thread copies 8 bytes of the 2KB tile per K-step
    const int srow = tid >> 3;                // 0..31
    const int sseg = tid & 7;                 // 0..7 (8B each)
    const unsigned short* gA = A + (size_t)(m0 + srow) * Kdim + sseg * 4;
    unsigned short* lA = &shA[srow * 32 + sseg * 4];

    const int brow_off = (half << 4) + l16;   // K index within 32-step for B

    for (int k0 = 0; k0 < Kdim; k0 += 32) {
        // ---- stage shared A tile (async DMA path on gfx1250) ----
        async_cp_b64(gA + k0, lA);
        wait_async0();
        __syncthreads();

        // ---- A fragments from LDS (16-bit A 16x32 layout) ----
        V16BF a[2];
        #pragma unroll
        for (int mt = 0; mt < 2; ++mt) {
            const unsigned int* ar =
                (const unsigned int*)&shA[(mt * 16 + l16) * 32];
            #pragma unroll
            for (int i = 0; i < 4; ++i) a[mt].u[i]     = ar[(half << 2) + i];
            #pragma unroll
            for (int i = 0; i < 4; ++i) a[mt].u[4 + i] = ar[8 + (half << 2) + i];
        }

        // ---- B fragments straight from global (lane = N column; elements over K, table order) ----
        V16BF b[4];
        const unsigned short* bp =
            Bw + (size_t)(k0 + (half << 3)) * Ndim + n0 + l16;
        #pragma unroll
        for (int nt = 0; nt < 4; ++nt)
            #pragma unroll
            for (int i = 0; i < 8; ++i) {
                const int kk = 2 * i + (i >= 4 ? 8 : 0);
                b[nt].u[i] = (unsigned int)bp[(size_t)kk * Ndim + nt * 16] |
                             ((unsigned int)bp[(size_t)(kk + 1) * Ndim + nt * 16] << 16);
            }

        if (k0 + 32 < Kdim)
            __builtin_prefetch(Bw + (size_t)(k0 + 32 + brow_off) * Ndim + n0, 0, 1);

        // ---- 8 WMMAs with register-level A/B reuse ----
        #pragma unroll
        for (int mt = 0; mt < 2; ++mt)
            #pragma unroll
            for (int nt = 0; nt < 4; ++nt)
                c[mt][nt] = __builtin_amdgcn_wmma_f32_16x16x32_bf16(
                    false, a[mt].v, false, b[nt].v, (short)0, c[mt][nt],
                    false, false);

        __syncthreads();   // protect shA before next K-step overwrite
    }

    // ---- epilogue: C/D layout -> global, optional residual add ----
    #pragma unroll
    for (int mt = 0; mt < 2; ++mt) {
        #pragma unroll
        for (int r = 0; r < 8; ++r) {
            const int m = m0 + mt * 16 + r + (half << 3);
            #pragma unroll
            for (int nt = 0; nt < 4; ++nt) {
                const int n = n0 + nt * 16 + l16;
                float v = c[mt][nt][r];
                if (Res) v += Res[(size_t)m * Ndim + n];
                Out[(size_t)m * Ndim + n] = v;
            }
        }
    }
}

// ---------------- q/k prep: l2norm per head + RoPE, in place on qkv ----------------
__global__ void qk_prep_k(float* __restrict__ qkv,
                          const float* __restrict__ cosT,
                          const float* __restrict__ sinT) {
    int i = blockIdx.x * blockDim.x + threadIdx.x;
    if (i >= Tc * Hc) return;
    const int t = i / Hc, h = i % Hc;
    const int s = t % Sc;
    for (int which = 0; which < 2; ++which) {
        float* p = qkv + (size_t)t * ThreeD + which * Dc + h * HDc;
        float v[HDc];
        float n2 = 0.f;
        #pragma unroll
        for (int d = 0; d < HDc; ++d) { v[d] = p[d]; n2 += v[d] * v[d]; }
        const float inv = 1.0f / fmaxf(sqrtf(n2), EPSc);
        #pragma unroll
        for (int j = 0; j < 32; ++j) {
            float x1 = v[j] * inv, x2 = v[j + 32] * inv;
            float cs = cosT[s * 32 + j], sn = sinT[s * 32 + j];
            p[j]      = x1 * cs + x2 * sn;
            p[j + 32] = -x1 * sn + x2 * cs;
        }
    }
}

// ---------------- attention: one thread per (b,h,q), online softmax ----------------
__global__ __launch_bounds__(64)
void attn_k(const float* __restrict__ qkv, const int* __restrict__ doc,
            float* __restrict__ out) {
    int i = blockIdx.x * blockDim.x + threadIdx.x;
    if (i >= Bc * Hc * Sc) return;
    const int q = i % Sc;
    const int h = (i / Sc) % Hc;
    const int b = i / (Sc * Hc);
    const int t = b * Sc + q;

    float qv[HDc];
    {
        const float* qp = qkv + (size_t)t * ThreeD + h * HDc;
        #pragma unroll
        for (int d = 0; d < HDc; ++d) qv[d] = qp[d];
    }
    const int qb = q / Ec, qr = q % Ec;
    const bool tok_q = (qr >= Kc);
    const int doc_q = doc[b * Lc + qb];

    float m = -3.4e38f, l = 0.f;
    float acc[HDc];
    #pragma unroll
    for (int d = 0; d < HDc; ++d) acc[d] = 0.f;

    const float scale = 0.125f;  // 1/sqrt(64)
    for (int k = 0; k < Sc; ++k) {
        const int kb = k / Ec, kr = k % Ec;
        const bool tok_k = (kr >= Kc);
        const bool same = (kb == qb);
        const bool prev = (qb == kb + 1);
        const int  kvbm = kb > 0 ? kb - 1 : 0;
        const bool dm = (doc_q == doc[b * Lc + kvbm]);
        const bool ok = (tok_q && same && (q >= k) && dm) ||
                        (tok_q && tok_k && prev && dm) ||
                        (!tok_q && !tok_k && same);
        if (!ok) continue;

        const float* kp = qkv + (size_t)(b * Sc + k) * ThreeD + Dc + h * HDc;
        float sd = 0.f;
        #pragma unroll
        for (int d = 0; d < HDc; ++d) sd += qv[d] * kp[d];
        sd *= scale;

        const float mnew = fmaxf(m, sd);
        const float corr = __expf(m - mnew);
        const float p    = __expf(sd - mnew);
        l = l * corr + p;
        const float* vp = qkv + (size_t)(b * Sc + k) * ThreeD + 2 * Dc + h * HDc;
        #pragma unroll
        for (int d = 0; d < HDc; ++d) acc[d] = acc[d] * corr + p * vp[d];
        m = mnew;
    }
    const float invl = (l > 0.f) ? (1.0f / l) : 0.f;
    float* op = out + (size_t)t * Dc + h * HDc;
    #pragma unroll
    for (int d = 0; d < HDc; ++d) op[d] = acc[d] * invl;
}

// ---------------- silu gate: up[T,2F] -> bf16 g[T,F] ----------------
__global__ void silugate_k(const float* __restrict__ up,
                           unsigned short* __restrict__ gbf, long n) {
    long i = (long)blockIdx.x * blockDim.x + threadIdx.x;
    if (i >= n) return;
    long t = i / Fc;
    int  j = (int)(i % Fc);
    float x1 = up[t * (size_t)TwoF + j];
    float x2 = up[t * (size_t)TwoF + Fc + j];
    float s = x1 / (1.0f + __expf(-x1));
    gbf[i] = f2bf(s * x2);
}

// ---------------- gather token rows (qr >= K) ----------------
__global__ void gather_out_k(const float* __restrict__ xs, float* __restrict__ out, long n) {
    long i = (long)blockIdx.x * blockDim.x + threadIdx.x;
    if (i >= n) return;
    long row = i / Dc;
    int  col = (int)(i % Dc);
    int  nn = (int)(row % Nc);
    long bl = row / Nc;
    int  ll = (int)(bl % Lc);
    int  bb = (int)(bl / Lc);
    size_t src = ((size_t)bb * Sc + (size_t)ll * Ec + Kc + nn) * Dc + col;
    out[i] = xs[src];
}

// ---------------- host-side orchestration ----------------
static inline size_t alignup(size_t x) { return (x + 255) & ~(size_t)255; }

extern "C" void kernel_launch(void* const* d_in, const int* in_sizes, int n_in,
                              void* d_out, int out_size, void* d_ws, size_t ws_size,
                              hipStream_t stream) {
    const float* x      = (const float*)d_in[0];
    const int*   doc    = (const int*)  d_in[1];
    const float* Wqkv   = (const float*)d_in[2];
    const float* Wo     = (const float*)d_in[3];
    const float* Wup    = (const float*)d_in[4];
    const float* Wdown  = (const float*)d_in[5];
    const float* g_attn = (const float*)d_in[6];
    const float* g_ffn  = (const float*)d_in[7];
    float* out = (float*)d_out;

    char* ws = (char*)d_ws;
    size_t off = 0;
    float* xs = (float*)(ws + off);            off += alignup((size_t)Tc * Dc * 4);
    float* r  = (float*)(ws + off);            off += alignup((size_t)Tc * Dc * 4);
    float* big = (float*)(ws + off);           off += alignup((size_t)Tc * TwoF * 4); // qkv / up alias
    float* attnout = (float*)(ws + off);       off += alignup((size_t)Tc * Dc * 4);
    unsigned short* abf = (unsigned short*)(ws + off); off += alignup((size_t)Tc * Fc * 2);
    unsigned short* wbf = (unsigned short*)(ws + off); off += alignup((size_t)Dc * TwoF * 2);
    float* cosT = (float*)(ws + off);          off += alignup((size_t)Sc * 32 * 4);
    float* sinT = (float*)(ws + off);          off += alignup((size_t)Sc * 32 * 4);
    float* qkv = big;
    float* up  = big;

    const long nXS = (long)Tc * Dc;
    copy_f32_k<<<(int)((nXS + 255) / 256), 256, 0, stream>>>(x, xs, nXS);
    rope_tables_k<<<(Sc * 32 + 255) / 256, 256, 0, stream>>>(cosT, sinT);

    for (int layer = 0; layer < NLc; ++layer) {
        // ---- attention block ----
        rmsnorm_bf16_k<<<Tc, 256, 0, stream>>>(xs, g_attn + layer * Dc, abf);
        long nqkvW = (long)Dc * ThreeD;
        conv_bf16_k<<<(int)((nqkvW + 255) / 256), 256, 0, stream>>>(
            Wqkv + (size_t)layer * Dc * ThreeD, wbf, nqkvW);
        gemm_wmma_bf16_k<<<dim3(ThreeD / 512, Tc / 32), 256, 0, stream>>>(
            abf, wbf, nullptr, qkv, Dc, ThreeD);

        qk_prep_k<<<(Tc * Hc + 255) / 256, 256, 0, stream>>>(qkv, cosT, sinT);
        attn_k<<<(Bc * Hc * Sc + 63) / 64, 64, 0, stream>>>(qkv, doc, attnout);

        conv_bf16_k<<<(int)((nXS + 255) / 256), 256, 0, stream>>>(attnout, abf, nXS);
        long nWo = (long)Dc * Dc;
        conv_bf16_k<<<(int)((nWo + 255) / 256), 256, 0, stream>>>(
            Wo + (size_t)layer * Dc * Dc, wbf, nWo);
        gemm_wmma_bf16_k<<<dim3(Dc / 512, Tc / 32), 256, 0, stream>>>(
            abf, wbf, xs, r, Dc, Dc);   // r = attnout@Wo + xs

        // ---- FFN block ----
        rmsnorm_bf16_k<<<Tc, 256, 0, stream>>>(r, g_ffn + layer * Dc, abf);
        long nWup = (long)Dc * TwoF;
        conv_bf16_k<<<(int)((nWup + 255) / 256), 256, 0, stream>>>(
            Wup + (size_t)layer * Dc * TwoF, wbf, nWup);
        gemm_wmma_bf16_k<<<dim3(TwoF / 512, Tc / 32), 256, 0, stream>>>(
            abf, wbf, nullptr, up, Dc, TwoF);

        long nG = (long)Tc * Fc;
        silugate_k<<<(int)((nG + 255) / 256), 256, 0, stream>>>(up, abf, nG);

        long nWd = (long)Fc * Dc;
        conv_bf16_k<<<(int)((nWd + 255) / 256), 256, 0, stream>>>(
            Wdown + (size_t)layer * Fc * Dc, wbf, nWd);
        gemm_wmma_bf16_k<<<dim3(Dc / 512, Tc / 32), 256, 0, stream>>>(
            abf, wbf, r, xs, Fc, Dc);   // xs = silu-gate@Wdown + r
    }

    const long nOut = (long)Bc * Lc * Nc * Dc;
    gather_out_k<<<(int)((nOut + 255) / 256), 256, 0, stream>>>(xs, out, nOut);
}
